// MultiHeadedSelfAttention_11622181503210
// MI455X (gfx1250) — matrix operand order repaired and checked by reading, not compiled, not run on hardware
//
#include <hip/hip_runtime.h>

#define DI __device__ __forceinline__

typedef __bf16 bf16;
typedef __attribute__((ext_vector_type(16))) __bf16 bf16x16;
typedef __attribute__((ext_vector_type(8)))  __bf16 bf16x8;
typedef __attribute__((ext_vector_type(8)))  float  f32x8;

constexpr int Bn = 4, Ln = 2048, Dn = 1024, Hn = 16, HDn = 64;
constexpr int Mn = Bn * Ln;            // 8192 rows of X
constexpr float NEG_BIG = -1.0e30f;

DI f32x8 zero8() {
  f32x8 z;
#pragma unroll
  for (int i = 0; i < 8; ++i) z[i] = 0.0f;
  return z;
}

// ---- WMMA fragment loaders (layouts per CDNA5 ISA 7.12.2, wave32) ----------
// A: 16x32 bf16, row-major source with leading dim ld.
// lane l: row = l&15 ; K = {h*8..h*8+7} U {16+h*8..16+h*8+7}, h = l>>4
DI bf16x16 load_a_frag(const bf16* __restrict__ base, int ld, int row0, int k0, int lane) {
  const int m = lane & 15, half = lane >> 4;
  const bf16* p = base + (size_t)(row0 + m) * ld + k0 + half * 8;
  bf16x8 lo = *reinterpret_cast<const bf16x8*>(p);
  bf16x8 hi = *reinterpret_cast<const bf16x8*>(p + 16);
  bf16x16 a;
#pragma unroll
  for (int j = 0; j < 8; ++j) { a[j] = lo[j]; a[8 + j] = hi[j]; }
  return a;
}

// B: 32x16 bf16 where B[k][n] is stored TRANSPOSED: bt[n*ld + k].
// lane l: col = l&15 ; K = {h*8..h*8+7} U {16+h*8..16+h*8+7}, h = l>>4
DI bf16x16 load_b_frag(const bf16* __restrict__ bt, int ld, int k0, int n0, int lane) {
  const int n = lane & 15, half = lane >> 4;
  const bf16* p = bt + (size_t)(n0 + n) * ld + k0 + half * 8;
  bf16x8 lo = *reinterpret_cast<const bf16x8*>(p);
  bf16x8 hi = *reinterpret_cast<const bf16x8*>(p + 16);
  bf16x16 b;
#pragma unroll
  for (int j = 0; j < 8; ++j) { b[j] = lo[j]; b[8 + j] = hi[j]; }
  return b;
}

DI f32x8 wmma_bf16(bf16x16 a, bf16x16 b, f32x8 c) {
  return __builtin_amdgcn_wmma_f32_16x16x32_bf16(false, a, false, b, (short)0, c, false, false);
}

// ---- Kernel 1: fp32 -> bf16 elementwise convert ----------------------------
__global__ __launch_bounds__(256) void k_cvt_x(const float* __restrict__ x,
                                               bf16* __restrict__ xb, int n) {
  int i = blockIdx.x * 256 + threadIdx.x;
  if (i < n) xb[i] = (bf16)x[i];
}

// ---- Kernel 2: transpose-convert the 4 weight matrices ---------------------
__global__ __launch_bounds__(256) void k_cvt_w(const float* __restrict__ wq, const float* __restrict__ wk,
                                               const float* __restrict__ wv, const float* __restrict__ wo,
                                               bf16* __restrict__ qt, bf16* __restrict__ kt,
                                               bf16* __restrict__ vt, bf16* __restrict__ ot) {
  const float* src = (blockIdx.y == 0) ? wq : (blockIdx.y == 1) ? wk : (blockIdx.y == 2) ? wv : wo;
  bf16* dst        = (blockIdx.y == 0) ? qt : (blockIdx.y == 1) ? kt : (blockIdx.y == 2) ? vt : ot;
  int i = blockIdx.x * 256 + threadIdx.x;     // grid sized to exactly Dn*Dn
  int n = i & (Dn - 1), k = i >> 10;
  dst[(size_t)n * Dn + k] = (bf16)src[(size_t)k * Dn + n];
}

// ---- Kernel 3: fused QKV projection GEMM (blockIdx.z selects Q/K/V) --------
// C[MxN] = Xb[MxK] * W[KxN]; wave tile 64x32 (4x2 WMMA), block tile 128x128
// (8 waves as 2 m-waves x 4 n-waves). 8 WMMA per 12 b128 loads per k-step.
__global__ __launch_bounds__(256) void k_gemm_qkv(
    const bf16* __restrict__ xb,
    const bf16* __restrict__ wqt, const bf16* __restrict__ wkt, const bf16* __restrict__ wvt,
    bf16* __restrict__ qo, bf16* __restrict__ ko, bf16* __restrict__ vto) {
  const int lane = threadIdx.x & 31, wid = threadIdx.x >> 5;
  const int m0 = blockIdx.x * 128 + (wid & 1) * 64;
  const int n0 = blockIdx.y * 128 + (wid >> 1) * 32;
  const int z = blockIdx.z;
  const bf16* wt = (z == 0) ? wqt : (z == 1) ? wkt : wvt;

  f32x8 acc[4][2];
#pragma unroll
  for (int i = 0; i < 4; ++i)
#pragma unroll
    for (int j = 0; j < 2; ++j) acc[i][j] = zero8();

  for (int kk = 0; kk < Dn; kk += 32) {
    if (kk + 32 < Dn) {   // L2 prefetch of next A strip (global_prefetch_b8)
      __builtin_prefetch(xb + (size_t)(m0 + (lane & 15)) * Dn + kk + 32, 0, 1);
    }
    bf16x16 a[4];
#pragma unroll
    for (int ti = 0; ti < 4; ++ti) a[ti] = load_a_frag(xb, Dn, m0 + ti * 16, kk, lane);
    bf16x16 b0 = load_b_frag(wt, Dn, kk, n0,      lane);
    bf16x16 b1 = load_b_frag(wt, Dn, kk, n0 + 16, lane);
#pragma unroll
    for (int ti = 0; ti < 4; ++ti) {
      acc[ti][0] = wmma_bf16(a[ti], b0, acc[ti][0]);
      acc[ti][1] = wmma_bf16(a[ti], b1, acc[ti][1]);
    }
  }

  const int half = lane >> 4, c = lane & 15;
#pragma unroll
  for (int ti = 0; ti < 4; ++ti)
#pragma unroll
    for (int tj = 0; tj < 2; ++tj)
#pragma unroll
      for (int r = 0; r < 8; ++r) {
        int gm = m0 + ti * 16 + r + half * 8;       // row in [0, B*L)
        int gn = n0 + tj * 16 + c;                  // col in [0, D)
        int b = gm >> 11, li = gm & (Ln - 1);
        int h = gn >> 6,  d  = gn & 63;
        bf16 v = (bf16)acc[ti][tj][r];
        if (z == 2)      vto[(((size_t)b * Hn + h) * HDn + d) * Ln + li] = v;  // V transposed
        else if (z == 0) qo[(((size_t)b * Hn + h) * Ln + li) * HDn + d] = v;
        else             ko[(((size_t)b * Hn + h) * Ln + li) * HDn + d] = v;
      }
}

// ---- Kernel 4: flash attention (causal + key-keep mask) --------------------
// One wave = 16 query rows; 32-key blocks; online softmax; all matmuls WMMA.
__global__ __launch_bounds__(256) void k_attn(
    const bf16* __restrict__ q, const bf16* __restrict__ k, const bf16* __restrict__ vt,
    const int* __restrict__ mask, bf16* __restrict__ ob) {
  __shared__ __align__(16) bf16 plds[8][16][32];   // per-wave P-tile scratch (8 KB)

  const int lane = threadIdx.x & 31, wid = threadIdx.x >> 5;
  const int bh = blockIdx.y;                       // b*H + h
  const int b = bh >> 4, h = bh & 15;
  const int q0 = blockIdx.x * 128 + wid * 16;      // this wave's query base
  const bf16* qh = q  + (size_t)bh * Ln * HDn;
  const bf16* kh = k  + (size_t)bh * Ln * HDn;
  const bf16* vh = vt + (size_t)bh * HDn * Ln;
  const int* mrow = mask + (size_t)b * Ln;
  const int half = lane >> 4, c = lane & 15;

  // Q fragments for K-steps 0..31 and 32..63 (hd = 64)
  bf16x16 aq0 = load_a_frag(qh, HDn, q0, 0,  lane);
  bf16x16 aq1 = load_a_frag(qh, HDn, q0, 32, lane);

  f32x8 o[4];
#pragma unroll
  for (int t = 0; t < 4; ++t) o[t] = zero8();
  float rm[8], rs[8];
#pragma unroll
  for (int r = 0; r < 8; ++r) { rm[r] = NEG_BIG; rs[r] = 0.0f; }

  const int kbMax = q0 + 16;                       // causal frontier (wave-uniform)
  for (int kb = 0; kb < kbMax; kb += 32) {
    // S = Q * K^T for 16 queries x 32 keys (two 16-col tiles, K-dim 64)
    f32x8 s[2];
#pragma unroll
    for (int nt = 0; nt < 2; ++nt) {
      f32x8 a = zero8();
      a = wmma_bf16(aq0, load_b_frag(kh, HDn, 0,  kb + nt * 16, lane), a);
      a = wmma_bf16(aq1, load_b_frag(kh, HDn, 32, kb + nt * 16, lane), a);
      s[nt] = a;
    }
    const bool keep0 = (mrow[kb + c] != 0);
    const bool keep1 = (mrow[kb + 16 + c] != 0);

#pragma unroll
    for (int r = 0; r < 8; ++r) {
      const int qi = q0 + r + half * 8;
      float s0 = s[0][r] * 0.125f;                 // 1/sqrt(64)
      float s1 = s[1][r] * 0.125f;
      s0 = (keep0 && (kb + c)      <= qi) ? s0 : NEG_BIG;
      s1 = (keep1 && (kb + 16 + c) <= qi) ? s1 : NEG_BIG;

      // row max across the 16 lanes holding this row (xor within half-wave)
      float mx = fmaxf(s0, s1);
#pragma unroll
      for (int msk = 1; msk < 16; msk <<= 1) mx = fmaxf(mx, __shfl_xor(mx, msk, 32));
      const float nm = fmaxf(rm[r], mx);
      const float alpha = __expf(rm[r] - nm);
      rm[r] = nm;

      const float p0 = __expf(s0 - nm);
      const float p1 = __expf(s1 - nm);
      float ps = p0 + p1;
#pragma unroll
      for (int msk = 1; msk < 16; msk <<= 1) ps += __shfl_xor(ps, msk, 32);
      rs[r] = rs[r] * alpha + ps;

#pragma unroll
      for (int t = 0; t < 4; ++t) o[t][r] *= alpha;

      const int m = r + half * 8;                  // C-layout row -> LDS row
      plds[wid][m][c]      = (bf16)p0;
      plds[wid][m][16 + c] = (bf16)p1;
    }

    // Re-layout P (C-fragment in LDS) into an A-fragment; same-wave LDS is in-order.
    bf16x16 ap;
    {
      bf16x8 lo = *reinterpret_cast<const bf16x8*>(&plds[wid][c][half * 8]);
      bf16x8 hi = *reinterpret_cast<const bf16x8*>(&plds[wid][c][16 + half * 8]);
#pragma unroll
      for (int j = 0; j < 8; ++j) { ap[j] = lo[j]; ap[8 + j] = hi[j]; }
    }

    // O += P * V  (B from transposed V: contiguous key runs)
#pragma unroll
    for (int nt = 0; nt < 4; ++nt)
      o[nt] = wmma_bf16(ap, load_b_frag(vh, Ln, kb, nt * 16, lane), o[nt]);
  }

  // normalize and store to [B][L][D] bf16 (attention output)
#pragma unroll
  for (int nt = 0; nt < 4; ++nt)
#pragma unroll
    for (int r = 0; r < 8; ++r) {
      const int qi = q0 + r + half * 8;
      const int d = nt * 16 + c;
      ob[((size_t)b * Ln + qi) * Dn + h * HDn + d] = (bf16)(o[nt][r] / rs[r]);
    }
}

// ---- Kernel 5: output projection, fp32 result ------------------------------
__global__ __launch_bounds__(256) void k_gemm_out(const bf16* __restrict__ ab,
                                                  const bf16* __restrict__ wot,
                                                  float* __restrict__ out) {
  const int lane = threadIdx.x & 31, wid = threadIdx.x >> 5;
  const int m0 = blockIdx.x * 128 + (wid & 1) * 64;
  const int n0 = blockIdx.y * 128 + (wid >> 1) * 32;

  f32x8 acc[4][2];
#pragma unroll
  for (int i = 0; i < 4; ++i)
#pragma unroll
    for (int j = 0; j < 2; ++j) acc[i][j] = zero8();

  for (int kk = 0; kk < Dn; kk += 32) {
    if (kk + 32 < Dn) {
      __builtin_prefetch(ab + (size_t)(m0 + (lane & 15)) * Dn + kk + 32, 0, 1);
    }
    bf16x16 a[4];
#pragma unroll
    for (int ti = 0; ti < 4; ++ti) a[ti] = load_a_frag(ab, Dn, m0 + ti * 16, kk, lane);
    bf16x16 b0 = load_b_frag(wot, Dn, kk, n0,      lane);
    bf16x16 b1 = load_b_frag(wot, Dn, kk, n0 + 16, lane);
#pragma unroll
    for (int ti = 0; ti < 4; ++ti) {
      acc[ti][0] = wmma_bf16(a[ti], b0, acc[ti][0]);
      acc[ti][1] = wmma_bf16(a[ti], b1, acc[ti][1]);
    }
  }

  const int half = lane >> 4, c = lane & 15;
#pragma unroll
  for (int ti = 0; ti < 4; ++ti)
#pragma unroll
    for (int tj = 0; tj < 2; ++tj)
#pragma unroll
      for (int r = 0; r < 8; ++r) {
        int gm = m0 + ti * 16 + r + half * 8;
        int gn = n0 + tj * 16 + c;
        out[(size_t)gm * Dn + gn] = acc[ti][tj][r];
      }
}

// ---- Host launcher ---------------------------------------------------------
extern "C" void kernel_launch(void* const* d_in, const int* in_sizes, int n_in,
                              void* d_out, int out_size, void* d_ws, size_t ws_size,
                              hipStream_t stream) {
  (void)in_sizes; (void)n_in; (void)out_size; (void)ws_size;
  const float* x    = (const float*)d_in[0];
  const int*   mask = (const int*)  d_in[1];
  const float* wq   = (const float*)d_in[2];
  const float* wk   = (const float*)d_in[3];
  const float* wv   = (const float*)d_in[4];
  const float* wo   = (const float*)d_in[5];

  char* ws = (char*)d_ws;
  size_t off = 0;
  bf16* xb  = (bf16*)(ws + off); off += (size_t)Mn * Dn * 2;   // 16.8 MB (reused as attn out)
  bf16* wqt = (bf16*)(ws + off); off += (size_t)Dn * Dn * 2;
  bf16* wkt = (bf16*)(ws + off); off += (size_t)Dn * Dn * 2;
  bf16* wvt = (bf16*)(ws + off); off += (size_t)Dn * Dn * 2;
  bf16* wot = (bf16*)(ws + off); off += (size_t)Dn * Dn * 2;   // +8 MB
  bf16* qb  = (bf16*)(ws + off); off += (size_t)Mn * Dn * 2;
  bf16* kb  = (bf16*)(ws + off); off += (size_t)Mn * Dn * 2;
  bf16* vtb = (bf16*)(ws + off); off += (size_t)Mn * Dn * 2;   // +50.3 MB  => ~75.5 MB total

  k_cvt_x<<<(Mn * Dn) / 256, 256, 0, stream>>>(x, xb, Mn * Dn);
  k_cvt_w<<<dim3((Dn * Dn) / 256, 4), 256, 0, stream>>>(wq, wk, wv, wo, wqt, wkt, wvt, wot);
  k_gemm_qkv<<<dim3(Mn / 128, Dn / 128, 3), 256, 0, stream>>>(xb, wqt, wkt, wvt, qb, kb, vtb);
  k_attn<<<dim3(Ln / 128, Bn * Hn), 256, 0, stream>>>(qb, kb, vtb, mask, /*ob=*/xb);
  k_gemm_out<<<dim3(Mn / 128, Dn / 128), 256, 0, stream>>>(xb, wot, (float*)d_out);
}
